// Llama_LLM_87024627351956
// MI455X (gfx1250) — matrix operand order repaired and checked by reading, not compiled, not run on hardware
//
#include <hip/hip_runtime.h>
#include <hip/hip_bf16.h>
#include <cstdint>

// Problem constants (match reference)
#define TV 32
#define TD 128
#define TH 8
#define THD 16
#define TB 8
#define TM 2048
#define MT (TB * TM)                 // 16384 total rows
#define NTOT ((size_t)MT * TD)       // 2,097,152 activations

typedef _Float16 half16 __attribute__((ext_vector_type(16)));
typedef float    float8 __attribute__((ext_vector_type(8)));

// ---------------------------------------------------------------- helpers
static __device__ __forceinline__ half16 h16_zero() {
  half16 z;
#pragma unroll
  for (int e = 0; e < 16; ++e) z[e] = (_Float16)0;
  return z;
}

static __device__ __forceinline__ float wave_reduce_sum(float v) {
#pragma unroll
  for (int off = 1; off < 32; off <<= 1) v += __shfl_xor(v, off, 32);
  return v;
}

// A operand: 16x32 f16 tile from row-major src (already offset to (row0,k0)), ld elems.
// lanes 0-15: M=lane, K = 0..15 (e0-15); lanes 16-31: K = 16..31
static __device__ __forceinline__ half16 load_a_f16(const _Float16* __restrict__ src, int ld) {
  const int lane = threadIdx.x & 31, hv = lane >> 4, r = lane & 15;
  const _Float16* p = src + (size_t)r * ld;
  half16 a;
#pragma unroll
  for (int e = 0; e < 16; ++e) a[e] = p[hv * 16 + e];
  return a;
}

// B operand: 32x16 f16 tile, B[k][n] = W[n0+n][k0+k]; W row-major, already offset to (n0,k0).
// lanes 0-15 hold K=0..15, lanes 16-31 hold K=16..31, n = lane&15.
static __device__ __forceinline__ half16 load_b_f16(const _Float16* __restrict__ w, int ld) {
  const int lane = threadIdx.x & 31, hv = lane >> 4, n = lane & 15;
  const _Float16* p = w + (size_t)n * ld + hv * 16;
  half16 b;
#pragma unroll
  for (int e = 0; e < 16; ++e) b[e] = p[e];
  return b;
}

// ---------------------------------------------------------------- f32 -> f16
__global__ void cvt_f32_f16(const float* __restrict__ src, _Float16* __restrict__ dst, int n) {
  int i = blockIdx.x * blockDim.x + threadIdx.x;
  if (i < n) dst[i] = (_Float16)src[i];
}

// ---------------------------------------------------------------- embedding + per-batch ssq
__global__ void embed_ssq(const int* __restrict__ idx, const float* __restrict__ emb,
                          float* __restrict__ x, float* __restrict__ ssq) {
  const int bm = blockIdx.x;            // row in [0, B*M)
  const int b  = bm / TM;
  const int d  = threadIdx.x;           // 128 threads
  float v = emb[(size_t)idx[bm] * TD + d];
  x[(size_t)bm * TD + d] = v;
  float s = wave_reduce_sum(v * v);
  if ((threadIdx.x & 31) == 0) atomicAdd(&ssq[b], s);
}

// ---------------------------------------------------------------- per-batch ssq of f32 tensor
__global__ void ssq_reduce(const float* __restrict__ x, float* __restrict__ ssq) {
  const size_t i = (size_t)blockIdx.x * blockDim.x + threadIdx.x;
  const int b = (int)(i / ((size_t)TM * TD));
  float v = x[i];
  float s = wave_reduce_sum(v * v);
  if ((threadIdx.x & 31) == 0) atomicAdd(&ssq[b], s);
}

// ---------------------------------------------------------------- batch RMS norm + scale
__global__ void rms_apply(const float* __restrict__ x, const float* __restrict__ ssq,
                          const float* __restrict__ scale, float* __restrict__ outF,
                          _Float16* __restrict__ outH) {
  const size_t i = (size_t)blockIdx.x * blockDim.x + threadIdx.x;
  const int b = (int)(i / ((size_t)TM * TD));
  const int m = (int)((i / TD) % TM);
  const int d = (int)(i % TD);
  const float inv = sqrtf((float)(TM * TD) / ssq[b]);     // x / fr == x * sqrt(M*D/ssq)
  float v = scale[(size_t)m * TD + d] * x[i] * inv;
  outF[i] = v;
  outH[i] = (_Float16)v;
}

// ---------------------------------------------------------------- RoPE (f32 in, f16 out)
__global__ void rope_apply(const float* __restrict__ in, _Float16* __restrict__ outH) {
  const size_t pi = (size_t)blockIdx.x * blockDim.x + threadIdx.x;  // pair index, NTOT/2 total
  const size_t row = pi / (TD / 2);
  const int m = (int)(row % TM);
  const int i = (int)(pi % (TD / 2));
  const float base = __powf(10000.0f, -2.0f * ((float)i - 1.0f) / (float)TD);
  float s, c;
  __sincosf((float)m * base, &s, &c);
  const float te = in[row * TD + 2 * i];
  const float to = in[row * TD + 2 * i + 1];
  outH[row * TD + 2 * i]     = (_Float16)( te * c + to * s);
  outH[row * TD + 2 * i + 1] = (_Float16)(-te * s + to * c);
}

// ---------------------------------------------------------------- generic WMMA GEMM
// out[row,n] = (RES? resid[row,n]:0) + maybe_relu( sum_d Xh[row,d]*Wh[n,d] + (BIAS? bias[n]:0) )
// All epilogue behavior is compile-time: branch-free inner code.
template <bool BIAS, bool RELU, bool RES, bool WF, bool WH>
__global__ __launch_bounds__(128) void wmma_gemm(
    const _Float16* __restrict__ Xh,   // MT x 128 row-major
    const _Float16* __restrict__ Wh,   // N  x 128 row-major
    const float* __restrict__ bias,    // N (if BIAS)
    const float* __restrict__ resid,   // MT x ldo (if RES)
    float* __restrict__ outF,          // MT x ldo (if WF)
    _Float16* __restrict__ outH,       // MT x ldo (if WH)
    int N, int ldo) {
  const int tile_m = blockIdx.x;
  const int wid    = threadIdx.x >> 5;
  const int nwaves = blockDim.x >> 5;
  const int lane   = threadIdx.x & 31, hv = lane >> 4, r16 = lane & 15;
  const int ntiles = N >> 4;
  const _Float16* xbase = Xh + (size_t)tile_m * 16 * TD;

  for (int nt = wid; nt < ntiles; nt += nwaves) {
    const _Float16* wbase = Wh + (size_t)nt * 16 * TD;
    float8 acc = {};
#pragma unroll
    for (int kc = 0; kc < TD; kc += 32) {
      if (kc + 32 < TD) __builtin_prefetch(wbase + kc + 32, 0, 3);   // global_prefetch_b8
      half16 a = load_a_f16(xbase + kc, TD);
      half16 b = load_b_f16(wbase + kc, TD);
      acc = __builtin_amdgcn_wmma_f32_16x16x32_f16(false, a, false, b,
                                                   (short)0, acc, false, false);
    }
    const int col  = nt * 16 + r16;
    const float bv = BIAS ? bias[col] : 0.0f;   // column is invariant across rr
#pragma unroll
    for (int rr = 0; rr < 8; ++rr) {
      const int row = tile_m * 16 + rr + hv * 8;
      float v = acc[rr] + bv;
      if (RELU) v = fmaxf(v, 0.0f);
      if (RES)  v += resid[(size_t)row * ldo + col];
      if (WF)   outF[(size_t)row * ldo + col] = v;
      if (WH)   outH[(size_t)row * ldo + col] = (_Float16)v;
    }
  }
}

// ---------------------------------------------------------------- flash attention (causal)
// grid = (M/16, H, B), block = 32 (one wave per 16-row query tile per head).
// Two key tiles per iteration: 2x QK^T WMMA (K=HD=16, zero-padded; unavoidable)
// feeding ONE fully-packed K=32 P.V WMMA (32 keys, no padding).
__global__ __launch_bounds__(32) void flash_attn(
    const _Float16* __restrict__ q, const _Float16* __restrict__ k,
    const _Float16* __restrict__ v, _Float16* __restrict__ o) {
  const int qt = blockIdx.x, h = blockIdx.y, b = blockIdx.z;
  const int lane = threadIdx.x, hv = lane >> 4, r16 = lane & 15;
  const int m0 = qt * 16;
  __shared__ _Float16 pl[16 * 32];   // 16 q-rows x 32 keys of probabilities

  // A operand for Q: rows = query, K = head-dim (16 real, 16 zero-pad)
  const _Float16* qb = q + ((size_t)b * TM + m0) * TD + h * THD;
  half16 aq = h16_zero();
#pragma unroll
  for (int e = 0; e < 16; ++e) if (hv == 0) aq[e] = qb[(size_t)r16 * TD + e];

  float m_st[8], l_st[8];
  float8 acc = {};
#pragma unroll
  for (int rr = 0; rr < 8; ++rr) { m_st[rr] = -1e30f; l_st[rr] = 0.0f; }

  for (int kt0 = 0; kt0 <= qt; kt0 += 2) {
    const int  k0   = kt0 * 16;
    const bool has2 = (kt0 + 1) <= qt;        // wave-uniform

    // ---- scores for key tile 0 (keys k0 .. k0+15)
    const _Float16* kb0 = k + ((size_t)b * TM + k0) * TD + h * THD;
    half16 bk0 = h16_zero();
    if (hv == 0) {
#pragma unroll
      for (int e = 0; e < 16; ++e) bk0[e] = kb0[(size_t)r16 * TD + e];
    }
    float8 z0 = {};
    float8 s0 = __builtin_amdgcn_wmma_f32_16x16x32_f16(false, aq, false, bk0,
                                                       (short)0, z0, false, false);
    // ---- scores for key tile 1 (keys k0+16 .. k0+31), if in range
    float8 s1 = {};
    if (has2) {                                // uniform branch: EXEC stays full
      const _Float16* kb1 = kb0 + 16 * TD;
      half16 bk1 = h16_zero();
      if (hv == 0) {
#pragma unroll
        for (int e = 0; e < 16; ++e) bk1[e] = kb1[(size_t)r16 * TD + e];
      }
      float8 z1 = {};
      s1 = __builtin_amdgcn_wmma_f32_16x16x32_f16(false, aq, false, bk1,
                                                  (short)0, z1, false, false);
    }

    // ---- joint streaming softmax update over both tiles
#pragma unroll
    for (int rr = 0; rr < 8; ++rr) {
      const int mq = m0 + rr + hv * 8;
      float sv0 = s0[rr] * 0.25f;                         // 1/sqrt(HD)
      if (k0 + r16 > mq) sv0 = -1e30f;                    // causal mask
      float sv1 = -1e30f;
      if (has2) {
        sv1 = s1[rr] * 0.25f;
        if (k0 + 16 + r16 > mq) sv1 = -1e30f;
      }
      float mx = fmaxf(sv0, sv1);                         // row max (16-lane half)
#pragma unroll
      for (int off = 1; off < 16; off <<= 1) mx = fmaxf(mx, __shfl_xor(mx, off, 32));
      const float mnew  = fmaxf(m_st[rr], mx);
      const float alpha = __expf(m_st[rr] - mnew);
      const float p0    = __expf(sv0 - mnew);
      const float p1    = has2 ? __expf(sv1 - mnew) : 0.0f;
      float rs = p0 + p1;
#pragma unroll
      for (int off = 1; off < 16; off <<= 1) rs += __shfl_xor(rs, off, 32);
      l_st[rr] = l_st[rr] * alpha + rs;
      m_st[rr] = mnew;
      acc[rr] *= alpha;
      const int prow = (rr + hv * 8) * 32;
      pl[prow + r16]      = (_Float16)p0;
      pl[prow + 16 + r16] = (_Float16)p1;
    }
    __syncthreads();

    // ---- A operand: full 16x32 probability tile (K = 32 keys, no padding)
    half16 ap;
#pragma unroll
    for (int e = 0; e < 16; ++e) ap[e] = pl[r16 * 32 + hv * 16 + e];

    // ---- B operand: V block 32x16 (lanes 0-15: keys k0..k0+15, lanes 16-31: k0+16..k0+31)
    const _Float16* vb = v + ((size_t)b * TM + k0) * TD + h * THD;
    half16 bv = h16_zero();
    if (hv == 0) {
#pragma unroll
      for (int e = 0; e < 16; ++e) bv[e] = vb[(size_t)e * TD + r16];
    } else if (has2) {
#pragma unroll
      for (int e = 0; e < 16; ++e) bv[e] = vb[(size_t)(16 + e) * TD + r16];
    }
    acc = __builtin_amdgcn_wmma_f32_16x16x32_f16(false, ap, false, bv,
                                                 (short)0, acc, false, false);
    __syncthreads();
  }
#pragma unroll
  for (int rr = 0; rr < 8; ++rr) {
    const int row = m0 + rr + hv * 8;
    o[((size_t)b * TM + row) * TD + h * THD + r16] = (_Float16)(acc[rr] / l_st[rr]);
  }
}

// ---------------------------------------------------------------- loss (log-softmax NLL mean)
__global__ void nll_loss(const float* __restrict__ logits, const int* __restrict__ tgt,
                         float* __restrict__ loss) {
  const int row  = blockIdx.x * (blockDim.x >> 5) + (threadIdx.x >> 5);
  const int lane = threadIdx.x & 31;
  const float* lp = logits + (size_t)row * TV;
  float vl = lp[lane];
  float mx = vl;
#pragma unroll
  for (int off = 1; off < 32; off <<= 1) mx = fmaxf(mx, __shfl_xor(mx, off, 32));
  float e = __expf(vl - mx);
#pragma unroll
  for (int off = 1; off < 32; off <<= 1) e += __shfl_xor(e, off, 32);
  const float logZ = mx + __logf(e);
  const float nll  = logZ - lp[tgt[row]];
  if (lane == 0) atomicAdd(loss, nll * (1.0f / (float)MT));
}

// ---------------------------------------------------------------- driver
extern "C" void kernel_launch(void* const* d_in, const int* in_sizes, int n_in,
                              void* d_out, int out_size, void* d_ws, size_t ws_size,
                              hipStream_t stream) {
  const int*   idx     = (const int*)  d_in[0];
  const int*   targets = (const int*)  d_in[1];
  const float* emb     = (const float*)d_in[2];
  const float* scale   = (const float*)d_in[3];
  const float* Wq      = (const float*)d_in[4];
  const float* Wk      = (const float*)d_in[5];
  const float* Wv      = (const float*)d_in[6];
  const float* in_w    = (const float*)d_in[7];
  const float* in_b    = (const float*)d_in[8];
  const float* outp_w  = (const float*)d_in[9];
  const float* outp_b  = (const float*)d_in[10];
  const float* lin_w   = (const float*)d_in[11];
  const float* lin_b   = (const float*)d_in[12];
  const float* out_w   = (const float*)d_in[13];
  const float* out_b   = (const float*)d_in[14];

  // ---- workspace bump allocator
  char* base = (char*)d_ws;
  size_t off = 0;
  auto alloc = [&](size_t bytes) { void* p = base + off; off = (off + bytes + 255) & ~(size_t)255; return p; };
  float*     ssq1   = (float*)    alloc(TB * sizeof(float));
  float*     ssq2   = (float*)    alloc(TB * sizeof(float));
  _Float16*  wq_h   = (_Float16*) alloc(TD * TD * 2);
  _Float16*  wk_h   = (_Float16*) alloc(TD * TD * 2);
  _Float16*  wv_h   = (_Float16*) alloc(TD * TD * 2);
  _Float16*  inw_h  = (_Float16*) alloc(3 * TD * TD * 2);
  _Float16*  outp_h = (_Float16*) alloc(TD * TD * 2);
  _Float16*  lin_h  = (_Float16*) alloc(TD * TD * 2);
  _Float16*  outw_h = (_Float16*) alloc(TV * TD * 2);
  float*     xA     = (float*)    alloc(NTOT * 4);   // x_emb, later x2
  float*     xn     = (float*)    alloc(NTOT * 4);
  _Float16*  xh     = (_Float16*) alloc(NTOT * 2);
  float*     tmpF   = (float*)    alloc(NTOT * 4);   // q/k/v pre-rope, later x4
  _Float16*  qrh    = (_Float16*) alloc(NTOT * 2);   // later x4h
  _Float16*  krh    = (_Float16*) alloc(NTOT * 2);
  _Float16*  vrh    = (_Float16*) alloc(NTOT * 2);
  _Float16*  qph    = (_Float16*) alloc(NTOT * 2);
  _Float16*  kph    = (_Float16*) alloc(NTOT * 2);
  _Float16*  vph    = (_Float16*) alloc(NTOT * 2);
  _Float16*  oh     = (_Float16*) alloc(NTOT * 2);
  float*     x3     = (float*)    alloc(NTOT * 4);
  _Float16*  x3h    = (_Float16*) alloc(NTOT * 2);
  (void)ws_size; (void)n_in; (void)in_sizes; (void)out_size;

  float* logits = (float*)d_out;                 // (B, M, V) row-major
  float* loss   = logits + (size_t)MT * TV;      // scalar at end

  hipMemsetAsync(ssq1, 0, TB * sizeof(float), stream);
  hipMemsetAsync(ssq2, 0, TB * sizeof(float), stream);
  hipMemsetAsync(loss, 0, sizeof(float), stream);

  // weights -> f16
  auto cvt = [&](const float* s, _Float16* d, int n) {
    cvt_f32_f16<<<(n + 255) / 256, 256, 0, stream>>>(s, d, n);
  };
  cvt(Wq, wq_h, TD * TD);        cvt(Wk, wk_h, TD * TD);        cvt(Wv, wv_h, TD * TD);
  cvt(in_w, inw_h, 3 * TD * TD); cvt(outp_w, outp_h, TD * TD);
  cvt(lin_w, lin_h, TD * TD);    cvt(out_w, outw_h, TV * TD);

  // x = emb[idx]; ssq1 per batch; xn = rms_norm(x)
  embed_ssq<<<MT, TD, 0, stream>>>(idx, emb, xA, ssq1);
  rms_apply<<<(int)(NTOT / 256), 256, 0, stream>>>(xA, ssq1, scale, xn, xh);

  const int GT = MT / 16;        // 1024 row tiles
  // Q/K/V projections (f32 out) + RoPE (f16 out)
  wmma_gemm<false,false,false,true,false><<<GT, 128, 0, stream>>>(xh, wq_h, nullptr, nullptr, tmpF, nullptr, TD, TD);
  rope_apply<<<(int)(NTOT / 2 / 256), 256, 0, stream>>>(tmpF, qrh);
  wmma_gemm<false,false,false,true,false><<<GT, 128, 0, stream>>>(xh, wk_h, nullptr, nullptr, tmpF, nullptr, TD, TD);
  rope_apply<<<(int)(NTOT / 2 / 256), 256, 0, stream>>>(tmpF, krh);
  wmma_gemm<false,false,false,true,false><<<GT, 128, 0, stream>>>(xh, wv_h, nullptr, nullptr, tmpF, nullptr, TD, TD);
  rope_apply<<<(int)(NTOT / 2 / 256), 256, 0, stream>>>(tmpF, vrh);

  // in_proj (qp/kp/vp), bias, f16 outputs
  wmma_gemm<true,false,false,false,true><<<GT, 128, 0, stream>>>(qrh, inw_h,               in_b,          nullptr, nullptr, qph, TD, TD);
  wmma_gemm<true,false,false,false,true><<<GT, 128, 0, stream>>>(krh, inw_h + TD * TD,     in_b + TD,     nullptr, nullptr, kph, TD, TD);
  wmma_gemm<true,false,false,false,true><<<GT, 128, 0, stream>>>(vrh, inw_h + 2 * TD * TD, in_b + 2 * TD, nullptr, nullptr, vph, TD, TD);

  // attention
  flash_attn<<<dim3(TM / 16, TH, TB), 32, 0, stream>>>(qph, kph, vph, oh);

  // out_proj + bias + residual (x2 = xn + proj), reuse xA as x2
  wmma_gemm<true,false,true,true,false><<<GT, 128, 0, stream>>>(oh, outp_h, outp_b, xn, xA, nullptr, TD, TD);

  // second rms_norm
  ssq_reduce<<<(int)(NTOT / 256), 256, 0, stream>>>(xA, ssq2);
  rms_apply<<<(int)(NTOT / 256), 256, 0, stream>>>(xA, ssq2, scale, x3, x3h);

  // FFN: x4 = x3 + relu(x3 @ lin.T + b); reuse tmpF as x4, qrh as x4h
  float* x4 = tmpF; _Float16* x4h = qrh;
  wmma_gemm<true,true,true,true,true><<<GT, 128, 0, stream>>>(x3h, lin_h, lin_b, x3, x4, x4h, TD, TD);

  // logits = x4 @ out_w.T + out_b  (N = 32)
  wmma_gemm<true,false,false,true,false><<<GT, 128, 0, stream>>>(x4h, outw_h, out_b, nullptr, logits, nullptr, TV, TV);

  // loss
  nll_loss<<<MT / 4, 128, 0, stream>>>(logits, targets, loss);
}
